// GPT_84524956385755
// MI455X (gfx1250) — matrix operand order repaired and checked by reading, not compiled, not run on hardware
//
#include <hip/hip_runtime.h>

// ---------------------------------------------------------------------------
// GPT forward (B=2,S=1024,D=1024,H=16,DH=64,HID=2816,V=50257,L=12)
// MI455X / gfx1250: wave32, v_wmma_f32_16x16x32_bf16, TDM-staged A tiles.
// HBM-bound on fp32 weight streaming (~822MB @ 23.3TB/s); weights converted
// fp32->bf16 in-kernel during LDS staging (no extra HBM pass).
// GEMM: 128x64 block tile, BK=64, 8 waves, 8 WMMA / K-iteration / wave.
// A tile (bf16 activations) staged via tensor_load_to_lds (TENSORcnt);
// B tile staged per-thread (needs f32->bf16 convert in flight).
// ---------------------------------------------------------------------------

#define Lc   12
#define Hc   16
#define Dc   1024
#define DHc  64
#define HIDc 2816
#define Vc   50257
#define Bc   2
#define Sc   1024
#define Mrows (Bc * Sc)   // 2048

#define BM 128
#define BN 64
#define BK 64

typedef __attribute__((ext_vector_type(16))) __bf16      v16bf;
typedef __attribute__((ext_vector_type(8)))  float       v8f;
typedef __attribute__((ext_vector_type(4))) unsigned int u32x4;
typedef __attribute__((ext_vector_type(8)))  int          i32x8;
typedef __attribute__((ext_vector_type(4)))  int          i32x4;
typedef __attribute__((ext_vector_type(4)))  float       f32x4;

#ifndef __has_builtin
#define __has_builtin(x) 0
#endif
#if __has_builtin(__builtin_amdgcn_tensor_load_to_lds) && \
    __has_builtin(__builtin_amdgcn_s_wait_tensorcnt)
#define USE_TDM 1
#else
#define USE_TDM 0
#endif

union Frag { v16bf v; u32x4 q[2]; };

__device__ __forceinline__ unsigned short f32_to_bf16(float f) {
    union { float f; unsigned int u; } v; v.f = f;
    unsigned int r = v.u + 0x7fffu + ((v.u >> 16) & 1u);   // RNE
    return (unsigned short)(r >> 16);
}

// ---------------------------------------------------------------------------
// Embedding: x[b,s,:] = wte[idx[b,s],:] + wpe[s,:]
// ---------------------------------------------------------------------------
__global__ __launch_bounds__(256)
void embed_kernel(const int* __restrict__ idx, const float* __restrict__ wte,
                  const float* __restrict__ wpe, float* __restrict__ x)
{
    int row = blockIdx.x;            // b*S + s
    int s   = row & (Sc - 1);
    int tok = idx[row];
    int t   = threadIdx.x;
#pragma unroll
    for (int i = 0; i < 4; ++i) {
        int d = t + 256 * i;
        x[(size_t)row * Dc + d] = wte[(size_t)tok * Dc + d] + wpe[(size_t)s * Dc + d];
    }
}

// ---------------------------------------------------------------------------
// RMSNorm -> bf16 (feeds WMMA A-operand)
// ---------------------------------------------------------------------------
__global__ __launch_bounds__(256)
void rmsnorm_kernel(const float* __restrict__ X, const float* __restrict__ gamma,
                    unsigned short* __restrict__ out)
{
    __shared__ float red[8];
    int row = blockIdx.x;
    int t   = threadIdx.x;
    const float* x = X + (size_t)row * Dc;
    float s = 0.f;
#pragma unroll
    for (int i = 0; i < 4; ++i) { float v = x[t + 256 * i]; s += v * v; }
#pragma unroll
    for (int off = 16; off > 0; off >>= 1) s += __shfl_down(s, off, 32);
    if ((t & 31) == 0) red[t >> 5] = s;
    __syncthreads();
    if (t < 8) {
        float v = red[t];
#pragma unroll
        for (int off = 4; off > 0; off >>= 1) v += __shfl_down(v, off, 8);
        if (t == 0) red[0] = v;
    }
    __syncthreads();
    float rms = rsqrtf(red[0] * (1.0f / (float)Dc) + 1e-12f);
#pragma unroll
    for (int i = 0; i < 4; ++i) {
        int d = t + 256 * i;
        out[(size_t)row * Dc + d] = f32_to_bf16(x[d] * rms * gamma[d]);
    }
}

// ---------------------------------------------------------------------------
// GEMM: C[M,N](f32) = A_bf16[M,K] @ W_f32[K,N] (+ optional residual R)
//   - 128x64 C tile per 256-thread block (8 waves), BK=64
//   - each wave: 32(M)x32(N) strip = c[2][2] accumulators, 8 WMMA / K-iter
//   - A tile: one TDM tensor_load_to_lds per iteration (wave 0 issues,
//     TENSORcnt wait, barrier publishes) -> zero VGPR round-trip
//   - B tile: column-segment per thread, coalesced b32 loads, bf16 cvt,
//     written fragment-major as 2x ds_store_b128
// ---------------------------------------------------------------------------
template<bool NGUARD>
__global__ __launch_bounds__(256)
void gemm_bf16_kernel(const unsigned short* __restrict__ A,
                      const float* __restrict__ W,
                      const float* __restrict__ R,     // residual or nullptr
                      float* __restrict__ C,
                      int M, int N, int K, int n_base)
{
    __shared__ unsigned short As[BM * BK];              // row-major [m][k], 16KB
    __shared__ unsigned short Bs[2 * 4 * 32 * 16];      // frag-major [ks][g][lane][i], 8KB

    const int tid  = threadIdx.x;
    const int lane = tid & 31;
    const int wave = tid >> 5;
    const int half = lane >> 4;
    const int r    = lane & 15;
    const int wm   = wave & 3;       // 4 M quarter-strips (32 rows each)
    const int wn   = wave >> 2;      // 2 N halves (32 cols each)
    const int m0   = blockIdx.y * BM;
    const int n0   = n_base + blockIdx.x * BN;

    // B staging role: thread -> (column n, k-segment of 16)
    const int bn    = tid & 63;          // 0..63
    const int kseg  = tid >> 6;          // 0..3
    const int bks   = kseg >> 1;         // which 32-K sub-tile
    const int blane = (kseg & 1) * 16 + (bn & 15);
    const int bg    = bn >> 4;           // 16-col group
    unsigned short* bdst = &Bs[(((bks * 4) + bg) * 32 + blane) * 16];
    const bool binb = (!NGUARD) || (n0 + bn < N);

#if USE_TDM
    // ---- TDM descriptor (constant parts). 2-D tensor: bf16 [M rows][K cols],
    //      tile = 64(K) x 128(M) -> row-major As[128][64] in LDS.
    const unsigned int lds_addr = (unsigned int)(size_t)(void*)&As[0];
    const unsigned long long gbase =
        (unsigned long long)(const void*)(A + (size_t)m0 * K);   // tile row 0
    u32x4 g0c;
    g0c.x = 1u;                       // count=1, user mode
    g0c.y = lds_addr;                 // bits 63:32  lds_addr
    i32x8 g1;
    g1.s0 = 0x00010000;               // wg_mask=0, data_size=1 (2 bytes)
    g1.s1 = (int)((unsigned)K << 16);             // tensor_dim0[15:0] @63:48
    g1.s2 = (int)(((unsigned)K >> 16) | ((unsigned)Mrows << 16)); // dim0 hi, dim1 lo
    g1.s3 = (int)(((unsigned)Mrows >> 16) | (64u << 16));         // dim1 hi, tile_dim0=64
    g1.s4 = (int)(unsigned)BM;        // tile_dim1=128, tile_dim2=0
    g1.s5 = (int)(unsigned)K;         // tensor_dim0_stride[31:0]
    g1.s6 = 0;                        // dim0_stride hi=0, dim1_stride lo=0
    g1.s7 = 0;
    i32x4 g2; g2.x = 0; g2.y = 0; g2.z = 0; g2.w = 0;   // 2-D: unused
    i32x4 g3; g3.x = 0; g3.y = 0; g3.z = 0; g3.w = 0;
    i32x8 g4; g4.s0 = 0; g4.s1 = 0; g4.s2 = 0; g4.s3 = 0;
              g4.s4 = 0; g4.s5 = 0; g4.s6 = 0; g4.s7 = 0;
#else
    // A staging role: thread -> (row, 32-elem half-row)
    const int arow = tid >> 1;
    const int acol = (tid & 1) * 32;
#endif

    v8f c[2][2] = {{{}, {}}, {{}, {}}};

    for (int k0 = 0; k0 < K; k0 += BK) {
#if USE_TDM
        // ---- stage A via Tensor Data Mover (one DMA per 16KB tile) ----
        if (wave == 0) {
            unsigned long long ga = gbase + (unsigned long long)k0 * 2ull;
            u32x4 g0 = g0c;
            g0.z = (unsigned int)ga;                               // addr[31:0]
            g0.w = (unsigned int)((ga >> 32) & 0x01FFFFFFull) | 0x80000000u; // addr hi | type=2
            __builtin_amdgcn_tensor_load_to_lds(g0, g1, g2, g3, g4, 0);
        }
#else
        {
            const u32x4* src = (const u32x4*)(A + (size_t)(m0 + arow) * K + k0 + acol);
            u32x4* dst = (u32x4*)&As[arow * BK + acol];
            dst[0] = src[0]; dst[1] = src[1]; dst[2] = src[2]; dst[3] = src[3];
        }
#endif
        // ---- stage B: column segment, f32 -> bf16, fragment-major -----
        {
            unsigned short tmp[16];
            if (binb) {
                const float* src = W + (size_t)(k0 + kseg * 16) * N + n0 + bn;
                if (k0 + BK < K)
                    __builtin_prefetch(src + (size_t)BK * N, 0, 1);
#pragma unroll
                for (int i = 0; i < 16; ++i)
                    tmp[i] = f32_to_bf16(src[(size_t)i * N]);
            } else {
#pragma unroll
                for (int i = 0; i < 16; ++i) tmp[i] = 0;
            }
            ((u32x4*)bdst)[0] = ((const u32x4*)tmp)[0];
            ((u32x4*)bdst)[1] = ((const u32x4*)tmp)[1];
        }
#if USE_TDM
        if (wave == 0) __builtin_amdgcn_s_wait_tensorcnt(0);
#endif
        __syncthreads();

        // ---- fragments (contiguous k = 16*half + s in A and B) ------------------
        Frag a[2][2], b[2][2];   // [ks][mi], [ks][gi]
#pragma unroll
        for (int ks = 0; ks < 2; ++ks) {
#pragma unroll
            for (int mi = 0; mi < 2; ++mi) {
                const unsigned short* ap =
                    &As[((wm * 2 + mi) * 16 + r) * BK + ks * 32 + half * 16];
                a[ks][mi].q[0] = *(const u32x4*)ap;           // K 0..7 / 16..23
                a[ks][mi].q[1] = *(const u32x4*)(ap + 8);     // K 8..15 / 24..31
            }
#pragma unroll
            for (int gi = 0; gi < 2; ++gi) {
                const unsigned short* bp =
                    &Bs[(((ks * 4) + wn * 2 + gi) * 32 + lane) * 16];
                b[ks][gi].q[0] = *(const u32x4*)bp;
                b[ks][gi].q[1] = *(const u32x4*)(bp + 8);
            }
        }
#pragma unroll
        for (int ks = 0; ks < 2; ++ks)
#pragma unroll
            for (int mi = 0; mi < 2; ++mi)
#pragma unroll
                for (int gi = 0; gi < 2; ++gi)
                    c[mi][gi] = __builtin_amdgcn_wmma_f32_16x16x32_bf16(
                        false, a[ks][mi].v, false, b[ks][gi].v,
                        (short)0, c[mi][gi], false, false);
        __syncthreads();
    }

    // ---- epilogue: C layout — VGPR j -> M=half*8+j, lane r -> N ----------
#pragma unroll
    for (int mi = 0; mi < 2; ++mi) {
#pragma unroll
        for (int gi = 0; gi < 2; ++gi) {
            int col = n0 + (wn * 2 + gi) * 16 + r;
            if (NGUARD && col >= N) continue;
#pragma unroll
            for (int j = 0; j < 8; ++j) {
                int row = m0 + (wm * 2 + mi) * 16 + half * 8 + j;
                float o = c[mi][gi][j];
                if (R) o += R[(size_t)row * N + col];
                C[(size_t)row * N + col] = o;
            }
        }
    }
}

// ---------------------------------------------------------------------------
// Flash-style attention: one block = (b, h, 64 q-rows); one thread per q-row.
// Online softmax, K/V tiles staged in LDS. Output bf16 (feeds proj GEMM).
// ---------------------------------------------------------------------------
__global__ __launch_bounds__(64)
void attention_kernel(const float* __restrict__ qkv, unsigned short* __restrict__ yb)
{
    __shared__ float Ks[64][64];
    __shared__ float Vs[64][64];
    const int qb = blockIdx.x, h = blockIdx.y, b = blockIdx.z;
    const int t  = threadIdx.x;
    const int q  = qb * 64 + t;
    const float scale = 0.125f;   // 1/sqrt(64)

    float qr[DHc], acc[DHc];
    const float* qrow = qkv + ((size_t)(b * Sc + q)) * (3 * Dc) + h * DHc;
#pragma unroll
    for (int d = 0; d < DHc; d += 4) {
        f32x4 v = *(const f32x4*)&qrow[d];
        qr[d] = v.x * scale; qr[d+1] = v.y * scale; qr[d+2] = v.z * scale; qr[d+3] = v.w * scale;
        acc[d] = 0.f; acc[d+1] = 0.f; acc[d+2] = 0.f; acc[d+3] = 0.f;
    }
    float mx = -1e30f, l = 0.f;

    for (int kb = 0; kb <= qb; ++kb) {
        const float* krow = qkv + ((size_t)(b * Sc + kb * 64 + t)) * (3 * Dc) + Dc + h * DHc;
        const float* vrow = krow + Dc;
        __syncthreads();
#pragma unroll
        for (int d = 0; d < DHc; d += 4) {
            *(f32x4*)&Ks[t][d] = *(const f32x4*)&krow[d];
            *(f32x4*)&Vs[t][d] = *(const f32x4*)&vrow[d];
        }
        __syncthreads();
        const int kmax = (kb == qb) ? (t + 1) : 64;   // causal mask
        for (int k = 0; k < kmax; ++k) {
            float s = 0.f;
#pragma unroll
            for (int d = 0; d < DHc; ++d) s = fmaf(qr[d], Ks[k][d], s);
            float nm = fmaxf(mx, s);
            float f  = __expf(mx - nm);
            float p  = __expf(s - nm);
            l = l * f + p;
#pragma unroll
            for (int d = 0; d < DHc; ++d) acc[d] = fmaf(acc[d], f, p * Vs[k][d]);
            mx = nm;
        }
    }
    float inv = 1.f / l;
    unsigned short* out = yb + ((size_t)(b * Sc + q)) * Dc + h * DHc;
#pragma unroll
    for (int d = 0; d < DHc; ++d) out[d] = f32_to_bf16(acc[d] * inv);
}

// ---------------------------------------------------------------------------
// SwiGLU: out_bf16 = silu(g) * u
// ---------------------------------------------------------------------------
__global__ __launch_bounds__(256)
void swiglu_kernel(const float* __restrict__ g, const float* __restrict__ u,
                   unsigned short* __restrict__ out, size_t n)
{
    size_t i = (size_t)blockIdx.x * 256 + threadIdx.x;
    if (i < n) {
        float gv = g[i];
        float sv = gv / (1.f + __expf(-gv));
        out[i] = f32_to_bf16(sv * u[i]);
    }
}

// ---------------------------------------------------------------------------
// Host orchestration
// ---------------------------------------------------------------------------
extern "C" void kernel_launch(void* const* d_in, const int* in_sizes, int n_in,
                              void* d_out, int out_size, void* d_ws, size_t ws_size,
                              hipStream_t stream)
{
    (void)in_sizes; (void)n_in; (void)out_size; (void)ws_size;

    const int*   idx   = (const int*)  d_in[0];
    const float* wte   = (const float*)d_in[1];
    const float* wpe   = (const float*)d_in[2];
    const float* g1    = (const float*)d_in[3];
    const float* Wqkv  = (const float*)d_in[4];
    const float* Wproj = (const float*)d_in[5];
    const float* g2    = (const float*)d_in[6];
    const float* Wg    = (const float*)d_in[7];
    const float* Wu    = (const float*)d_in[8];
    const float* Wd    = (const float*)d_in[9];
    const float* gf    = (const float*)d_in[10];
    const float* Wlm   = (const float*)d_in[11];

    // workspace partition (all segment sizes are multiples of 256B)
    char* p = (char*)d_ws;
    float*          x    = (float*)p;          p += (size_t)Mrows * Dc   * 4;  // 8 MB
    unsigned short* hb   = (unsigned short*)p; p += (size_t)Mrows * Dc   * 2;  // 4 MB
    float*          qkvb = (float*)p;          p += (size_t)Mrows * 3*Dc * 4;  // 24 MB
    unsigned short* yb   = (unsigned short*)p; p += (size_t)Mrows * Dc   * 2;  // 4 MB
    float*          gbuf = (float*)p;          p += (size_t)Mrows * HIDc * 4;  // 22 MB
    float*          ubuf = (float*)p;          p += (size_t)Mrows * HIDc * 4;  // 22 MB
    unsigned short* ab   = (unsigned short*)p; p += (size_t)Mrows * HIDc * 2;  // 11 MB

    dim3 blk256(256), blk64(64);
    const int MB = Mrows / BM;   // 16

    embed_kernel<<<Mrows, blk256, 0, stream>>>(idx, wte, wpe, x);

    for (int l = 0; l < Lc; ++l) {
        // ---- attention block ----
        rmsnorm_kernel<<<Mrows, blk256, 0, stream>>>(x, g1 + (size_t)l * Dc, hb);
        gemm_bf16_kernel<false><<<dim3(3 * Dc / BN, MB), blk256, 0, stream>>>(
            hb, Wqkv + (size_t)l * Dc * 3 * Dc, nullptr, qkvb, Mrows, 3 * Dc, Dc, 0);
        attention_kernel<<<dim3(Sc / 64, Hc, Bc), blk64, 0, stream>>>(qkvb, yb);
        gemm_bf16_kernel<false><<<dim3(Dc / BN, MB), blk256, 0, stream>>>(
            yb, Wproj + (size_t)l * Dc * Dc, x, x, Mrows, Dc, Dc, 0);

        // ---- SwiGLU FFN block ----
        rmsnorm_kernel<<<Mrows, blk256, 0, stream>>>(x, g2 + (size_t)l * Dc, hb);
        gemm_bf16_kernel<false><<<dim3(HIDc / BN, MB), blk256, 0, stream>>>(
            hb, Wg + (size_t)l * Dc * HIDc, nullptr, gbuf, Mrows, HIDc, Dc, 0);
        gemm_bf16_kernel<false><<<dim3(HIDc / BN, MB), blk256, 0, stream>>>(
            hb, Wu + (size_t)l * Dc * HIDc, nullptr, ubuf, Mrows, HIDc, Dc, 0);
        size_t nsw = (size_t)Mrows * HIDc;
        swiglu_kernel<<<(unsigned)((nsw + 255) / 256), blk256, 0, stream>>>(gbuf, ubuf, ab, nsw);
        gemm_bf16_kernel<false><<<dim3(Dc / BN, MB), blk256, 0, stream>>>(
            ab, Wd + (size_t)l * HIDc * Dc, x, x, Mrows, Dc, HIDc, 0);
    }

    // ---- final norm + LM head (full blocks unguarded + 1 ragged edge block)
    rmsnorm_kernel<<<Mrows, blk256, 0, stream>>>(x, gf, hb);
    const int vfull = Vc / BN;                    // 785 full 64-col blocks
    gemm_bf16_kernel<false><<<dim3(vfull, MB), blk256, 0, stream>>>(
        hb, Wlm, nullptr, (float*)d_out, Mrows, Vc, Dc, 0);
    gemm_bf16_kernel<true><<<dim3(1, MB), blk256, 0, stream>>>(
        hb, Wlm, nullptr, (float*)d_out, Mrows, Vc, Dc, vfull * BN);
}
